// BeamSearchDecoder_57045755625958
// MI455X (gfx1250) — matrix operand order repaired and checked by reading, not compiled, not run on hardware
//
#include <hip/hip_runtime.h>
#include <hip/hip_bf16.h>
#include <math.h>

#define VOCAB  32000
#define HID    512
#define NLAYER 2
#define SRCLEN 64
#define BEAMS  3
#define MAXLEN 50
#define GATE_N (4 * HID)   // 2048 gate outputs per layer
#define KXH    (2 * HID)   // 1024 = concat(x, h) contraction length

typedef _Float16 half_t;
typedef __attribute__((ext_vector_type(16))) _Float16 v16h;
typedef __attribute__((ext_vector_type(8)))  _Float16 v8h;
typedef __attribute__((ext_vector_type(8)))  float    v8f;

__device__ __forceinline__ float sigmoidf_(float x) { return 1.0f / (1.0f + __expf(-x)); }

// ---------------------------------------------------------------------------
// One-time: W_out (V x H, f32) -> f16. 32 MB result stays L2-resident across
// all 50 decode steps (192 MB L2).
// ---------------------------------------------------------------------------
__global__ void cvt_f16_kernel(const float* __restrict__ w, half_t* __restrict__ o, int n) {
    int i = blockIdx.x * blockDim.x + threadIdx.x;
    int stride = gridDim.x * blockDim.x;
    for (; i < n; i += stride) o[i] = (half_t)w[i];
}

// ---------------------------------------------------------------------------
// One-time: pack per-layer [Wih | Whh] into row-major [L][4H][1024] f16 so the
// two gate matvecs become a single K=1024 WMMA GEMM per layer-step.
// ---------------------------------------------------------------------------
__global__ void pack_gates_kernel(const float* __restrict__ Wih, const float* __restrict__ Whh,
                                  half_t* __restrict__ out) {
    size_t n = (size_t)NLAYER * GATE_N * KXH;
    size_t i = (size_t)blockIdx.x * blockDim.x + threadIdx.x;
    size_t stride = (size_t)gridDim.x * blockDim.x;
    for (; i < n; i += stride) {
        size_t k  = i % KXH;
        size_t lj = i / KXH;                 // l*4H + j
        float v = (k < HID) ? Wih[lj * HID + k] : Whh[lj * HID + (k - HID)];
        out[i] = (half_t)v;
    }
}

// ---------------------------------------------------------------------------
// Init beam state: tokens = BOS, scores = 0, backtrack filled with BOS.
// ---------------------------------------------------------------------------
__global__ void init_kernel(const int* __restrict__ bos_p, int* __restrict__ tokens,
                            float* __restrict__ scores, int* __restrict__ backtrack) {
    int bos = bos_p[0];
    int tid = threadIdx.x;
    if (tid < BEAMS) { tokens[tid] = bos; scores[tid] = 0.0f; }
    for (int i = tid; i < BEAMS * MAXLEN; i += blockDim.x) backtrack[i] = bos;
}

// ---------------------------------------------------------------------------
// Encoder: one persistent workgroup (8 waves), 64 steps x 2 layers.
// Gate matvec z = [x|h] (16x1024, row 0 live) x Wcat^T via WMMA:
//   A fragments ds_load_b128 from a zero-padded LDS tile,
//   B fragments global_load_b128 from the packed f16 weights.
// ---------------------------------------------------------------------------
__global__ __launch_bounds__(256)
void enc_kernel_wmma(const int* __restrict__ seq, const float* __restrict__ emb,
                     const half_t* __restrict__ Wcat, const float* __restrict__ bias,
                     float* __restrict__ enc_out,
                     float* __restrict__ h_state, float* __restrict__ c_state) {
    __shared__ __align__(16) half_t xh[16][KXH];      // 32 KB A-tile (row 0 live)
    __shared__ float zrow[GATE_N];                    // 8 KB
    __shared__ float hs[NLAYER][HID];
    __shared__ float cs[NLAYER][HID];
    int tid = threadIdx.x;
    int waveId = tid >> 5, lane = tid & 31;
    int r   = lane & 15;
    int hiA = (lane >> 4) * 8;    // A halves: lanes 16-31 take K+8 / K+24 subranges
    int hiB = (lane >> 4) * 16;   // unused: B uses the same K+8 / K+24 subranges as A
    for (int i = tid; i < 16 * KXH; i += 256) ((half_t*)xh)[i] = (half_t)0.0f;
    for (int j = tid; j < HID; j += 256) {
        hs[0][j] = 0.0f; hs[1][j] = 0.0f; cs[0][j] = 0.0f; cs[1][j] = 0.0f;
    }
    __syncthreads();
    for (int s = 0; s < SRCLEN; ++s) {
        int tok = seq[s];
        for (int l = 0; l < NLAYER; ++l) {
            // pack row 0 = [x | h_l] in f16
            for (int j = tid; j < HID; j += 256) {
                float x = (l == 0) ? emb[(size_t)tok * HID + j] : hs[0][j];
                xh[0][j]       = (half_t)x;
                xh[0][HID + j] = (half_t)hs[l][j];
            }
            __syncthreads();
            const half_t* Wl = Wcat + (size_t)l * GATE_N * KXH;
            for (int tile = waveId; tile < GATE_N / 16; tile += 8) {
                int n0 = tile * 16;
                const half_t* bRow = Wl + (size_t)(n0 + r) * KXH;
                v8f acc = {};
                for (int k0 = 0; k0 < KXH; k0 += 32) {
                    union { v16h v; v8h h[2]; } ua, ub;
                    ua.h[0] = *(const v8h*)(&xh[r][k0 + hiA]);
                    ua.h[1] = *(const v8h*)(&xh[r][k0 + 16 + hiA]);
                    ub.h[0] = *(const v8h*)(bRow + k0 + hiA);
                    ub.h[1] = *(const v8h*)(bRow + k0 + 16 + hiA);
                    acc = __builtin_amdgcn_wmma_f32_16x16x32_f16(
                        false, ua.v, false, ub.v, (short)0, acc, false, false);
                }
                if (lane < 16) zrow[n0 + r] = acc[0];   // M=0 row lives in VGPR0, lanes 0-15
            }
            __syncthreads();
            const float* bl = bias + l * GATE_N;
            for (int j = tid; j < HID; j += 256) {
                float ig = sigmoidf_(zrow[j]           + bl[j]);
                float fg = sigmoidf_(zrow[HID + j]     + bl[HID + j]);
                float gg = tanhf    (zrow[2 * HID + j] + bl[2 * HID + j]);
                float og = sigmoidf_(zrow[3 * HID + j] + bl[3 * HID + j]);
                float cc = fg * cs[l][j] + ig * gg;
                float hh = og * tanhf(cc);
                cs[l][j] = cc; hs[l][j] = hh;
            }
            __syncthreads();
        }
        for (int j = tid; j < HID; j += 256) enc_out[s * HID + j] = hs[1][j];
        __syncthreads();
    }
    for (int l = 0; l < NLAYER; ++l)
        for (int k = 0; k < BEAMS; ++k)
            for (int j = tid; j < HID; j += 256) {
                h_state[(l * BEAMS + k) * HID + j] = hs[l][j];
                c_state[(l * BEAMS + k) * HID + j] = cs[l][j];
            }
}

// ---------------------------------------------------------------------------
// Decoder LSTM, both layers, all beams in one workgroup: beam rows 0..2 of a
// 16x1024 A-tile, z (3x2048) via WMMA, scalar gates. 62 KB LDS.
// ---------------------------------------------------------------------------
__global__ __launch_bounds__(256)
void lstm_dec_wmma(const int* __restrict__ tokens, const float* __restrict__ emb,
                   const half_t* __restrict__ Wcat, const float* __restrict__ bias,
                   const float* __restrict__ h_in, const float* __restrict__ c_in,
                   float* __restrict__ h_out, float* __restrict__ c_out) {
    __shared__ __align__(16) half_t xh[16][KXH];      // 32 KB
    __shared__ float zz[BEAMS][GATE_N];               // 24 KB
    __shared__ float hcur[BEAMS][HID];                // 6 KB
    int tid = threadIdx.x;
    int waveId = tid >> 5, lane = tid & 31;
    int r   = lane & 15;
    int hiA = (lane >> 4) * 8;
    int hiB = (lane >> 4) * 16;
    for (int i = tid; i < 16 * KXH; i += 256) ((half_t*)xh)[i] = (half_t)0.0f;
    __syncthreads();
    for (int l = 0; l < NLAYER; ++l) {
        for (int idx = tid; idx < BEAMS * HID; idx += 256) {
            int b = idx / HID, j = idx % HID;
            float x = (l == 0) ? emb[(size_t)tokens[b] * HID + j] : hcur[b][j];
            xh[b][j]       = (half_t)x;
            xh[b][HID + j] = (half_t)h_in[(l * BEAMS + b) * HID + j];
        }
        __syncthreads();
        const half_t* Wl = Wcat + (size_t)l * GATE_N * KXH;
        for (int tile = waveId; tile < GATE_N / 16; tile += 8) {
            int n0 = tile * 16;
            const half_t* bRow = Wl + (size_t)(n0 + r) * KXH;
            v8f acc = {};
            for (int k0 = 0; k0 < KXH; k0 += 32) {
                union { v16h v; v8h h[2]; } ua, ub;
                ua.h[0] = *(const v8h*)(&xh[r][k0 + hiA]);
                ua.h[1] = *(const v8h*)(&xh[r][k0 + 16 + hiA]);
                ub.h[0] = *(const v8h*)(bRow + k0 + hiA);
                ub.h[1] = *(const v8h*)(bRow + k0 + 16 + hiA);
                acc = __builtin_amdgcn_wmma_f32_16x16x32_f16(
                    false, ua.v, false, ub.v, (short)0, acc, false, false);
            }
            if (lane < 16) {   // rows M=0,1,2 live in acc[0..2], lanes 0-15
                zz[0][n0 + r] = acc[0];
                zz[1][n0 + r] = acc[1];
                zz[2][n0 + r] = acc[2];
            }
        }
        __syncthreads();
        const float* bl = bias + l * GATE_N;
        for (int idx = tid; idx < BEAMS * HID; idx += 256) {
            int b = idx / HID, j = idx % HID;
            float ig = sigmoidf_(zz[b][j]           + bl[j]);
            float fg = sigmoidf_(zz[b][HID + j]     + bl[HID + j]);
            float gg = tanhf    (zz[b][2 * HID + j] + bl[2 * HID + j]);
            float og = sigmoidf_(zz[b][3 * HID + j] + bl[3 * HID + j]);
            float cc = fg * c_in[(l * BEAMS + b) * HID + j] + ig * gg;
            float hh = og * tanhf(cc);
            c_out[(l * BEAMS + b) * HID + j] = cc;
            h_out[(l * BEAMS + b) * HID + j] = hh;
            hcur[b][j] = hh;
        }
        __syncthreads();
    }
}

// ---------------------------------------------------------------------------
// Attention + tanh(W_c . concat(x,ctx)) per beam; writes the zero-padded f16
// A-tile directly (block BEAMS zeroes the pad rows).
// ---------------------------------------------------------------------------
__global__ __launch_bounds__(256)
void attn_feat_kernel(const float* __restrict__ h_top, const float* __restrict__ enc_out,
                      const float* __restrict__ Wc, half_t* __restrict__ featPad) {
    int b = blockIdx.x;
    int tid = threadIdx.x;
    if (b >= BEAMS) {
        for (int idx = tid; idx < (16 - BEAMS) * HID; idx += blockDim.x)
            featPad[BEAMS * HID + idx] = (half_t)0.0f;
        return;
    }
    __shared__ float xs[HID];
    __shared__ float attn[SRCLEN];
    __shared__ float ctx[HID];
    for (int j = tid; j < HID; j += blockDim.x) xs[j] = h_top[b * HID + j];
    __syncthreads();
    if (tid < SRCLEN) {
        const float* er = enc_out + tid * HID;
        float acc = 0.0f;
        for (int k = 0; k < HID; ++k) acc += er[k] * xs[k];
        attn[tid] = acc;
    }
    __syncthreads();
    if (tid == 0) {
        float m = attn[0];
        for (int s = 1; s < SRCLEN; ++s) m = fmaxf(m, attn[s]);
        float sum = 0.0f;
        for (int s = 0; s < SRCLEN; ++s) { float e = __expf(attn[s] - m); attn[s] = e; sum += e; }
        float inv = 1.0f / sum;
        for (int s = 0; s < SRCLEN; ++s) attn[s] *= inv;
    }
    __syncthreads();
    for (int j = tid; j < HID; j += blockDim.x) {
        float acc = 0.0f;
        for (int s = 0; s < SRCLEN; ++s) acc += attn[s] * enc_out[s * HID + j];
        ctx[j] = acc;
    }
    __syncthreads();
    for (int j = tid; j < HID; j += blockDim.x) {
        const float* wr = Wc + (size_t)j * 2 * HID;
        float acc = 0.0f;
        for (int k = 0; k < HID; ++k) acc += wr[k] * xs[k];
        for (int k = 0; k < HID; ++k) acc += wr[HID + k] * ctx[k];
        featPad[b * HID + j] = (half_t)tanhf(acc);
    }
}

// ---------------------------------------------------------------------------
// Output projection: logits[b][v] = feat[b] . W_out[v] + b_out[v].
// One wave per 16-vocab tile, K=512 in chunks of 32 via v_wmma_f32_16x16x32_f16.
// ---------------------------------------------------------------------------
__global__ __launch_bounds__(256)
void out_gemm_kernel(const half_t* __restrict__ A, const half_t* __restrict__ W,
                     const float* __restrict__ bias, float* __restrict__ logits) {
    int wave = blockIdx.x * (blockDim.x / 32) + (threadIdx.x / 32);
    int lane = threadIdx.x & 31;
    int v0 = wave * 16;
    if (v0 >= VOCAB) return;           // whole-wave uniform; EXEC all-1s around WMMA
    int r   = lane & 15;
    int hiA = (lane >> 4) * 8;
    int hiB = (lane >> 4) * 16;
    const half_t* aRow = A + r * HID;
    const half_t* bRow = W + (size_t)(v0 + r) * HID;
    v8f acc = {};
    for (int k0 = 0; k0 < HID; k0 += 32) {
        union { v16h v; v8h h[2]; } ua, ub;
        ua.h[0] = *(const v8h*)(aRow + k0 + hiA);
        ua.h[1] = *(const v8h*)(aRow + k0 + 16 + hiA);
        ub.h[0] = *(const v8h*)(bRow + k0 + hiA);
        ub.h[1] = *(const v8h*)(bRow + k0 + 16 + hiA);
        acc = __builtin_amdgcn_wmma_f32_16x16x32_f16(
            false, ua.v, false, ub.v, (short)0, acc, false, false);
    }
    float bv = bias[v0 + r];
#pragma unroll
    for (int rr = 0; rr < 8; ++rr) {
        int M = (lane < 16) ? rr : rr + 8;
        if (M < BEAMS) logits[(size_t)M * VOCAB + v0 + r] = acc[rr] + bv;
    }
}

// ---------------------------------------------------------------------------
// Per-beam: LSE over 32000 logits, deterministic top-3 (first-occurrence ties).
// ---------------------------------------------------------------------------
__global__ __launch_bounds__(256)
void topk_kernel(const float* __restrict__ logits, float* __restrict__ topv,
                 int* __restrict__ topi) {
    int b = blockIdx.x;
    int tid = threadIdx.x;
    const float* row = logits + (size_t)b * VOCAB;
    __shared__ float red[256];
    float m = -1e30f;
    for (int v = tid; v < VOCAB; v += 256) m = fmaxf(m, row[v]);
    red[tid] = m; __syncthreads();
    for (int s = 128; s > 0; s >>= 1) { if (tid < s) red[tid] = fmaxf(red[tid], red[tid + s]); __syncthreads(); }
    m = red[0]; __syncthreads();
    float sum = 0.0f;
    for (int v = tid; v < VOCAB; v += 256) sum += __expf(row[v] - m);
    red[tid] = sum; __syncthreads();
    for (int s = 128; s > 0; s >>= 1) { if (tid < s) red[tid] += red[tid + s]; __syncthreads(); }
    float lse = m + __logf(red[0]);
    __syncthreads();
    float lv[3] = { -1e30f, -1e30f, -1e30f };
    int   li[3] = { VOCAB, VOCAB, VOCAB };
    for (int v = tid; v < VOCAB; v += 256) {
        float x = row[v];
        if (x > lv[0])      { lv[2]=lv[1]; li[2]=li[1]; lv[1]=lv[0]; li[1]=li[0]; lv[0]=x; li[0]=v; }
        else if (x > lv[1]) { lv[2]=lv[1]; li[2]=li[1]; lv[1]=x; li[1]=v; }
        else if (x > lv[2]) { lv[2]=x; li[2]=v; }
    }
    __shared__ float cv[256 * 3];
    __shared__ int   ci[256 * 3];
    for (int k = 0; k < 3; ++k) { cv[tid * 3 + k] = lv[k]; ci[tid * 3 + k] = li[k]; }
    __syncthreads();
    if (tid == 0) {
        int chosen[3];
        for (int k = 0; k < 3; ++k) {
            float bv = -1e30f; int bi = VOCAB;
            for (int j = 0; j < 256 * 3; ++j) {
                int idx = ci[j];
                if (idx >= VOCAB) continue;
                bool used = false;
                for (int q = 0; q < k; ++q) if (chosen[q] == idx) used = true;
                if (used) continue;
                float v = cv[j];
                if (v > bv || (v == bv && idx < bi)) { bv = v; bi = idx; }
            }
            chosen[k] = bi;
            topv[b * 3 + k] = bv - lse;
            topi[b * 3 + k] = bi;
        }
    }
}

// ---------------------------------------------------------------------------
// Beam bookkeeping: KxK -> top K, reorder h/c, backtrack gather via LDS.
// ---------------------------------------------------------------------------
__global__ __launch_bounds__(256)
void beam_update_kernel(const float* __restrict__ topv, const int* __restrict__ topi,
                        float* __restrict__ scores, int* __restrict__ tokens,
                        const float* __restrict__ h_cand, const float* __restrict__ c_cand,
                        float* __restrict__ h_state, float* __restrict__ c_state,
                        int* __restrict__ backtrack, int t, int isFirst) {
    int tid = threadIdx.x;
    __shared__ float oldSc[BEAMS];
    __shared__ int   oldTok[BEAMS];
    __shared__ int   orig[BEAMS];
    __shared__ int   newTok[BEAMS];
    __shared__ float newSc[BEAMS];
    __shared__ int   btRows[BEAMS][MAXLEN];
    if (tid == 0) {
        for (int k = 0; k < BEAMS; ++k) { oldSc[k] = scores[k]; oldTok[k] = tokens[k]; }
        if (isFirst) {
            for (int k = 0; k < BEAMS; ++k) { newSc[k] = topv[k]; newTok[k] = topi[k]; orig[k] = 0; }
        } else {
            float cum[BEAMS * BEAMS];
            for (int f = 0; f < BEAMS * BEAMS; ++f) cum[f] = topv[f] + oldSc[f / BEAMS];
            int chosen[BEAMS];
            for (int k = 0; k < BEAMS; ++k) {
                float bv = -1e30f; int bf = 0;
                for (int f = 0; f < BEAMS * BEAMS; ++f) {
                    bool used = false;
                    for (int q = 0; q < k; ++q) if (chosen[q] == f) used = true;
                    if (used) continue;
                    if (cum[f] > bv) { bv = cum[f]; bf = f; }   // strict > : lowest flat wins
                }
                chosen[k] = bf; newSc[k] = bv; newTok[k] = topi[bf]; orig[k] = bf / BEAMS;
            }
        }
    }
    __syncthreads();
    if (!isFirst) {
        for (int idx = tid; idx < BEAMS * MAXLEN; idx += blockDim.x) {
            int k = idx / MAXLEN, j = idx % MAXLEN;
            btRows[k][j] = backtrack[orig[k] * MAXLEN + j];
        }
    }
    __syncthreads();
    for (int idx = tid; idx < NLAYER * BEAMS * HID; idx += blockDim.x) {
        int l = idx / (BEAMS * HID);
        int rem = idx % (BEAMS * HID);
        int k = rem / HID, j = rem % HID;
        h_state[(l * BEAMS + k) * HID + j] = h_cand[(l * BEAMS + orig[k]) * HID + j];
        c_state[(l * BEAMS + k) * HID + j] = c_cand[(l * BEAMS + orig[k]) * HID + j];
    }
    if (!isFirst) {
        for (int idx = tid; idx < BEAMS * MAXLEN; idx += blockDim.x) {
            int k = idx / MAXLEN, j = idx % MAXLEN;
            backtrack[k * MAXLEN + j] = (j == t + 1) ? oldTok[orig[k]] : btRows[k][j];
        }
    }
    if (tid < BEAMS) { scores[tid] = newSc[tid]; tokens[tid] = newTok[tid]; }
}

// ---------------------------------------------------------------------------
// Final: argmax over scores (first max) and emit that backtrack row.
// ---------------------------------------------------------------------------
__global__ void final_kernel(const float* __restrict__ scores, const int* __restrict__ backtrack,
                             int* __restrict__ out) {
    __shared__ int best;
    if (threadIdx.x == 0) {
        int b = 0; float bv = scores[0];
        for (int k = 1; k < BEAMS; ++k) if (scores[k] > bv) { bv = scores[k]; b = k; }
        best = b;
    }
    __syncthreads();
    for (int j = threadIdx.x; j < MAXLEN; j += blockDim.x) out[j] = backtrack[best * MAXLEN + j];
}

// ---------------------------------------------------------------------------
extern "C" void kernel_launch(void* const* d_in, const int* in_sizes, int n_in,
                              void* d_out, int out_size, void* d_ws, size_t ws_size,
                              hipStream_t stream) {
    const int*   input_seq = (const int*)d_in[0];
    const float* emb_enc   = (const float*)d_in[1];
    const float* Wih_enc   = (const float*)d_in[2];
    const float* Whh_enc   = (const float*)d_in[3];
    const float* b_enc     = (const float*)d_in[4];
    const float* emb_dec   = (const float*)d_in[5];
    const float* Wih_dec   = (const float*)d_in[6];
    const float* Whh_dec   = (const float*)d_in[7];
    const float* b_dec     = (const float*)d_in[8];
    const float* W_c       = (const float*)d_in[9];
    const float* W_out     = (const float*)d_in[10];
    const float* b_out     = (const float*)d_in[11];
    const int*   bos       = (const int*)d_in[14];
    int* out = (int*)d_out;

    char* ws = (char*)d_ws;
    size_t off = 0;
    auto alloc = [&](size_t bytes) -> void* {
        void* p = ws + off;
        off = (off + bytes + 255) & ~(size_t)255;
        return p;
    };
    half_t* WoutH    = (half_t*)alloc((size_t)VOCAB * HID * sizeof(half_t));
    half_t* WencH    = (half_t*)alloc((size_t)NLAYER * GATE_N * KXH * sizeof(half_t));
    half_t* WdecH    = (half_t*)alloc((size_t)NLAYER * GATE_N * KXH * sizeof(half_t));
    float*  enc_out  = (float*)alloc((size_t)SRCLEN * HID * sizeof(float));
    float*  h_state  = (float*)alloc((size_t)NLAYER * BEAMS * HID * sizeof(float));
    float*  c_state  = (float*)alloc((size_t)NLAYER * BEAMS * HID * sizeof(float));
    float*  h_cand   = (float*)alloc((size_t)NLAYER * BEAMS * HID * sizeof(float));
    float*  c_cand   = (float*)alloc((size_t)NLAYER * BEAMS * HID * sizeof(float));
    half_t* featPad  = (half_t*)alloc((size_t)16 * HID * sizeof(half_t));
    float*  logits   = (float*)alloc((size_t)BEAMS * VOCAB * sizeof(float));
    float*  topv     = (float*)alloc((size_t)BEAMS * BEAMS * sizeof(float));
    int*    topi     = (int*)alloc((size_t)BEAMS * BEAMS * sizeof(int));
    float*  scores   = (float*)alloc((size_t)BEAMS * sizeof(float));
    int*    tokens   = (int*)alloc((size_t)BEAMS * sizeof(int));
    int*    backtrack= (int*)alloc((size_t)BEAMS * MAXLEN * sizeof(int));

    // One-time weight conversion / packing (f32 -> f16, gate concat).
    cvt_f16_kernel<<<4096, 256, 0, stream>>>(W_out, WoutH, VOCAB * HID);
    pack_gates_kernel<<<2048, 256, 0, stream>>>(Wih_enc, Whh_enc, WencH);
    pack_gates_kernel<<<2048, 256, 0, stream>>>(Wih_dec, Whh_dec, WdecH);

    init_kernel<<<1, 256, 0, stream>>>(bos, tokens, scores, backtrack);

    // Encoder: persistent single workgroup, WMMA gate GEMMs.
    enc_kernel_wmma<<<1, 256, 0, stream>>>(input_seq, emb_enc, WencH, b_enc,
                                           enc_out, h_state, c_state);

    auto decode_step = [&](int t, int isFirst) {
        lstm_dec_wmma<<<1, 256, 0, stream>>>(tokens, emb_dec, WdecH, b_dec,
                                             h_state, c_state, h_cand, c_cand);
        attn_feat_kernel<<<4, 256, 0, stream>>>(h_cand + (size_t)(NLAYER - 1) * BEAMS * HID,
                                                enc_out, W_c, featPad);
        out_gemm_kernel<<<(VOCAB / 16) / 8, 256, 0, stream>>>(featPad, WoutH, b_out, logits);
        topk_kernel<<<BEAMS, 256, 0, stream>>>(logits, topv, topi);
        beam_update_kernel<<<1, 256, 0, stream>>>(topv, topi, scores, tokens,
                                                  h_cand, c_cand, h_state, c_state,
                                                  backtrack, t, isFirst);
    };

    decode_step(-1, 1);                                  // first step (beam 0 only)
    for (int t = 0; t < MAXLEN - 1; ++t) decode_step(t, 0);

    final_kernel<<<1, 64, 0, stream>>>(scores, backtrack, out);
}
